// RoomTransformerModel_86234353369313
// MI455X (gfx1250) — matrix operand order repaired and checked by reading, not compiled, not run on hardware
//
#include <hip/hip_runtime.h>
#include <hip/hip_bf16.h>

typedef __bf16 bf16;
typedef __attribute__((ext_vector_type(16))) __bf16 bf16x16;
typedef __attribute__((ext_vector_type(8)))  __bf16 bf16x8;
typedef __attribute__((ext_vector_type(8)))  float  f32x8;
typedef __attribute__((ext_vector_type(4)))  int    i32x4v;

#define NB    32
#define RR    512
#define SS    513
#define LL    6
#define MROWS (NB*SS)        // 16416 real rows
#define MPAD  16512          // 129 * 128, padded for 128-row block tiles

// Detect gfx1250 async global->LDS builtins (toolchain-dependent)
#if defined(__has_builtin)
#  if __has_builtin(__builtin_amdgcn_global_load_async_to_lds_b128)
#    define HAVE_ASYNC_LDS 1
#  endif
#endif

#if defined(HAVE_ASYNC_LDS)
typedef __attribute__((address_space(1))) i32x4v* gas1_p;   // global int4*
typedef __attribute__((address_space(3))) i32x4v* las3_p;   // LDS int4*
__device__ inline void async_copy16(const void* g, void* l) {
  __builtin_amdgcn_global_load_async_to_lds_b128((gas1_p)(void*)g, (las3_p)l, 0, 0);
}
#endif

__device__ inline void wait_async0() {
#if defined(__has_builtin) && __has_builtin(__builtin_amdgcn_s_wait_asynccnt)
  __builtin_amdgcn_s_wait_asynccnt(0);
#else
  asm volatile("s_wait_asynccnt 0" ::: "memory");
#endif
}

// ---------------------------------------------------------------------------
// fp32 -> bf16 conversion (weights; every launch, deterministic)
// ---------------------------------------------------------------------------
__global__ void cvt_f32_bf16(const float* __restrict__ s, bf16* __restrict__ d, size_t n) {
  size_t i = (size_t)blockIdx.x * blockDim.x + threadIdx.x;
  size_t stride = (size_t)gridDim.x * blockDim.x;
  for (; i < n; i += stride) d[i] = (bf16)s[i];
}

// ---------------------------------------------------------------------------
// WMMA GEMM: C[M,N] = A[M,K] @ W[N,K]^T (+gelu) (+resid)
// 256 threads = 8 waves (4 M x 2 N), block tile 128x128, wave tile 32x64.
// K%32==0, N%128==0, M%128==0. LDS double-buffered, fragment-swizzled tiles.
// ---------------------------------------------------------------------------
#define BK 32

__global__ __launch_bounds__(256)
void gemm_wmma_bf16(const bf16* __restrict__ A, const bf16* __restrict__ W,
                    const float* __restrict__ resid,
                    float* __restrict__ Cf, bf16* __restrict__ Cb,
                    int M, int K, int N, int Mvalid, int gelu_flag) {
  __shared__ __align__(32) bf16 lA[2][128 * BK];   // fragment-major: [mt*32+lane][16]
  __shared__ __align__(32) bf16 lB[2][128 * BK];   // fragment-major: [nt*32+lane][16]

  const int tid  = threadIdx.x;
  const int lane = tid & 31;
  const int wv   = tid >> 5;
  const int wm   = wv >> 1;          // 0..3
  const int wn   = wv & 1;           // 0..1
  const int lr   = lane & 15;
  const int lh   = lane >> 4;
  const int m0g  = blockIdx.x * 128;
  const int n0g  = blockIdx.y * 128;

  // --- staging geometry (fixed per thread) ---
  // B: one 32B chunk: row n (0..127), k-half h (0..1)
  const int bn = tid >> 1, bh = tid & 1;
  const bf16* pB = W + (size_t)(n0g + bn) * K + bh * 16;
  const int dstB = (((bn >> 4) * 32) + (bn & 15) + 16 * bh) * 16;
  // A: two 16B chunks: rows m, m+64; k-group g (0..3)
  const int am = tid >> 2, ag = tid & 3;
  const bf16* pA0 = A + (size_t)(m0g + am) * K + 8 * ag;
  const bf16* pA1 = A + (size_t)(m0g + am + 64) * K + 8 * ag;
  const int dstA0 = (((am >> 4) * 32) + (am & 15) + 16 * (ag >> 1)) * 16 + (ag & 1) * 8;
  const int dstA1 = ((((am + 64) >> 4) * 32) + (am & 15) + 16 * (ag >> 1)) * 16 + (ag & 1) * 8;

  // --- fragment read indices (element units) ---
  const int iA0 = ((wm * 2 + 0) * 32 + lane) * 16;
  const int iA1 = ((wm * 2 + 1) * 32 + lane) * 16;
  int iB[4];
#pragma unroll
  for (int t = 0; t < 4; ++t) iB[t] = ((wn * 4 + t) * 32 + lane) * 16;

  f32x8 acc[2][4];
#pragma unroll
  for (int s = 0; s < 2; ++s)
#pragma unroll
    for (int t = 0; t < 4; ++t) acc[s][t] = (f32x8){0.f,0.f,0.f,0.f,0.f,0.f,0.f,0.f};

#if defined(HAVE_ASYNC_LDS)
  auto stage_async = [&](int buf, int k0) {
    async_copy16(pB + k0,      &lB[buf][dstB]);
    async_copy16(pB + k0 + 8,  &lB[buf][dstB + 8]);
    async_copy16(pA0 + k0,     &lA[buf][dstA0]);
    async_copy16(pA1 + k0,     &lA[buf][dstA1]);
  };
  stage_async(0, 0);
  wait_async0();
  __syncthreads();
#else
  bf16x8 rb0, rb1, ra0, ra1;
  auto ld_stage = [&](int k0) {
    rb0 = *(const bf16x8*)(pB + k0);
    rb1 = *(const bf16x8*)(pB + k0 + 8);
    ra0 = *(const bf16x8*)(pA0 + k0);
    ra1 = *(const bf16x8*)(pA1 + k0);
  };
  auto st_stage = [&](int buf) {
    *(bf16x8*)&lB[buf][dstB]     = rb0;
    *(bf16x8*)&lB[buf][dstB + 8] = rb1;
    *(bf16x8*)&lA[buf][dstA0]    = ra0;
    *(bf16x8*)&lA[buf][dstA1]    = ra1;
  };
  ld_stage(0);
  st_stage(0);
  __syncthreads();
#endif

  int cur = 0;
  for (int k0 = 0; k0 < K; k0 += BK) {
    const bool more = (k0 + BK) < K;
#if defined(HAVE_ASYNC_LDS)
    if (more) stage_async(cur ^ 1, k0 + BK);   // prev reads of cur^1 finished at last barrier
#else
    if (more) ld_stage(k0 + BK);
#endif
    // compute on current buffer: 2 A frags x 4 B frags = 8 WMMAs
    bf16x16 a0 = *(const bf16x16*)&lA[cur][iA0];
    bf16x16 a1 = *(const bf16x16*)&lA[cur][iA1];
#pragma unroll
    for (int t = 0; t < 4; ++t) {
      bf16x16 b = *(const bf16x16*)&lB[cur][iB[t]];
      acc[0][t] = __builtin_amdgcn_wmma_f32_16x16x32_bf16(false, a0, false, b, (short)0, acc[0][t], false, false);
      acc[1][t] = __builtin_amdgcn_wmma_f32_16x16x32_bf16(false, a1, false, b, (short)0, acc[1][t], false, false);
    }
#if defined(HAVE_ASYNC_LDS)
    if (more) wait_async0();
#else
    if (more) st_stage(cur ^ 1);
#endif
    __syncthreads();
    cur ^= 1;
  }

  // epilogue
#pragma unroll
  for (int s = 0; s < 2; ++s) {
#pragma unroll
    for (int t = 0; t < 4; ++t) {
      const int col = n0g + wn * 64 + t * 16 + lr;
#pragma unroll
      for (int v = 0; v < 8; ++v) {
        const int row = m0g + wm * 32 + s * 16 + v + 8 * lh;
        if (row < Mvalid) {
          size_t idx = (size_t)row * N + col;
          float x = acc[s][t][v];
          if (gelu_flag) x = 0.5f * x * (1.0f + erff(x * 0.70710678118f));
          if (resid) x += resid[idx];
          if (Cf) Cf[idx] = x;
          if (Cb) Cb[idx] = (bf16)x;
        }
      }
    }
  }
}

// ---------------------------------------------------------------------------
// Flash attention, one wave per (qtile, head, batch). GQA: head h -> group h/4.
// KV fused buffer: row stride 256, cols [0,128)=K, [128,256)=V.
// ---------------------------------------------------------------------------
__device__ inline float hredmax16(float x) {
#pragma unroll
  for (int m = 1; m < 16; m <<= 1) x = fmaxf(x, __shfl_xor(x, m, 32));
  return x;
}
__device__ inline float hredsum16(float x) {
#pragma unroll
  for (int m = 1; m < 16; m <<= 1) x += __shfl_xor(x, m, 32);
  return x;
}

__global__ __launch_bounds__(32)
void attn_flash(const bf16* __restrict__ Qb, const bf16* __restrict__ KVb,
                bf16* __restrict__ Ab) {
  __shared__ bf16 Pl[16 * 34];
  __shared__ bf16 Vt[64 * 34];

  const int lane = threadIdx.x;
  const int lr = lane & 15, lh = lane >> 4;
  const int qt = blockIdx.x;
  const int h  = blockIdx.y;
  const int b  = blockIdx.z;
  const int g  = h >> 2;
  const size_t base = (size_t)b * SS;
  const int qcol = g * 256 + (h & 3) * 64;
  const int kcol = g * 64;

  int qrow = qt * 16 + lr; if (qrow > 512) qrow = 512;
  const bf16* qp = Qb + (base + qrow) * 512 + qcol;
  bf16x16 qa0, qa1;
  *((bf16x8*)&qa0)       = *((const bf16x8*)(qp + lh * 8));
  *(((bf16x8*)&qa0) + 1) = *((const bf16x8*)(qp + lh * 8 + 16));
  *((bf16x8*)&qa1)       = *((const bf16x8*)(qp + 32 + lh * 8));
  *(((bf16x8*)&qa1) + 1) = *((const bf16x8*)(qp + 32 + lh * 8 + 16));

  f32x8 o[4];
#pragma unroll
  for (int t = 0; t < 4; ++t) o[t] = (f32x8){0.f,0.f,0.f,0.f,0.f,0.f,0.f,0.f};
  float mrow[8], lrow[8];
#pragma unroll
  for (int v = 0; v < 8; ++v) { mrow[v] = -1e30f; lrow[v] = 0.f; }

  for (int c = 0; c < 17; ++c) {
    int key = c * 32 + lane; if (key > 512) key = 512;
    const bf16* vp = KVb + (base + key) * 256 + 128 + kcol;
    bf16x8 vv[8];
#pragma unroll
    for (int i = 0; i < 8; ++i) vv[i] = *((const bf16x8*)(vp + i * 8));
#pragma unroll
    for (int i = 0; i < 8; ++i)
#pragma unroll
      for (int j2 = 0; j2 < 8; ++j2)
        Vt[(i * 8 + j2) * 34 + lane] = vv[i][j2];

    f32x8 d[2];
#pragma unroll
    for (int j = 0; j < 2; ++j) {
      int keyn = c * 32 + j * 16 + lr;
      int keync = keyn > 512 ? 512 : keyn;
      const bf16* kp = KVb + (base + keync) * 256 + kcol + lh * 8;
      bf16x16 kb0, kb1;
      *((bf16x8*)&kb0)       = *((const bf16x8*)(kp));
      *(((bf16x8*)&kb0) + 1) = *((const bf16x8*)(kp + 16));
      *((bf16x8*)&kb1)       = *((const bf16x8*)(kp + 32));
      *(((bf16x8*)&kb1) + 1) = *((const bf16x8*)(kp + 48));
      f32x8 dd = (f32x8){0.f,0.f,0.f,0.f,0.f,0.f,0.f,0.f};
      dd = __builtin_amdgcn_wmma_f32_16x16x32_bf16(false, qa0, false, kb0, (short)0, dd, false, false);
      dd = __builtin_amdgcn_wmma_f32_16x16x32_bf16(false, qa1, false, kb1, (short)0, dd, false, false);
      float msk = (keyn <= 512) ? 0.f : -1e30f;
#pragma unroll
      for (int v = 0; v < 8; ++v) dd[v] = dd[v] * 0.125f + msk;
      d[j] = dd;
    }
    __syncthreads();

#pragma unroll
    for (int v = 0; v < 8; ++v) {
      float rm = hredmax16(fmaxf(d[0][v], d[1][v]));
      float mnew = fmaxf(mrow[v], rm);
      float fac = __expf(mrow[v] - mnew);
      mrow[v] = mnew;
      float p0 = __expf(d[0][v] - mnew);
      float p1 = __expf(d[1][v] - mnew);
      lrow[v] = lrow[v] * fac + hredsum16(p0 + p1);
#pragma unroll
      for (int t = 0; t < 4; ++t) o[t][v] *= fac;
      int prow = v + lh * 8;
      Pl[prow * 34 + lr]      = (bf16)p0;
      Pl[prow * 34 + 16 + lr] = (bf16)p1;
    }
    __syncthreads();

    bf16x16 pa;
#pragma unroll
    for (int i = 0; i < 8; ++i) pa[i]     = Pl[lr * 34 + lh * 8 + i];
#pragma unroll
    for (int i = 0; i < 8; ++i) pa[8 + i] = Pl[lr * 34 + 16 + lh * 8 + i];
#pragma unroll
    for (int t = 0; t < 4; ++t) {
      bf16x16 bv;
#pragma unroll
      for (int i = 0; i < 16; ++i) bv[i] = Vt[(t * 16 + lr) * 34 + lh * 8 + (i & 7) + (i >> 3) * 16];
      o[t] = __builtin_amdgcn_wmma_f32_16x16x32_bf16(false, pa, false, bv, (short)0, o[t], false, false);
    }
    __syncthreads();
  }

#pragma unroll
  for (int v = 0; v < 8; ++v) {
    int r = qt * 16 + v + lh * 8;
    if (r < SS) {
      float inv = 1.0f / lrow[v];
#pragma unroll
      for (int t = 0; t < 4; ++t)
        Ab[(base + r) * 512 + h * 64 + t * 16 + lr] = (bf16)(o[t][v] * inv);
    }
  }
}

// ---------------------------------------------------------------------------
// Embedding / pooling / layernorm helper kernels
// ---------------------------------------------------------------------------
__global__ __launch_bounds__(256)
void gemb_kernel(const unsigned char* __restrict__ mask, const float* __restrict__ steps,
                 const float* __restrict__ temp, const float* __restrict__ mc,
                 const float* __restrict__ gw, const float* __restrict__ gb,
                 float* __restrict__ Xf, bf16* __restrict__ Xb) {
  int b = blockIdx.x;
  for (int e = threadIdx.x; e < 512; e += 256) {
    const float* wrow = gw + (size_t)e * 515;
    float acc = gb[e];
    for (int r = 0; r < 512; ++r) acc += mask[b * 512 + r] ? wrow[r] : 0.0f;
    acc += wrow[512] * (steps[b] * (1.0f / 512.0f));
    acc += wrow[513] * logf(temp[b]);
    acc += wrow[514] * mc[b];
    size_t idx = (size_t)b * SS * 512 + e;
    Xf[idx] = acc; Xb[idx] = (bf16)acc;
  }
}

__global__ __launch_bounds__(128)
void embed_kernel(const unsigned char* __restrict__ mask,
                  const int* __restrict__ posx, const int* __restrict__ posy,
                  const int* __restrict__ halfx, const int* __restrict__ halfy,
                  const float* __restrict__ pex, const float* __restrict__ pey,
                  const float* __restrict__ remb, const float* __restrict__ uemb,
                  float* __restrict__ Xf, bf16* __restrict__ Xb) {
  int blk = blockIdx.x;
  int b = blk >> 9, r = blk & 511;
  size_t row = (size_t)b * SS + 1 + r;
  bool m = mask[b * 512 + r] != 0;
  int ix = posx[b * 512 + r] + halfx[r];
  int iy = posy[b * 512 + r] + halfy[r];
  for (int c = threadIdx.x; c < 512; c += 128) {
    float v = m ? (pex[(size_t)ix * 512 + c] + pey[(size_t)iy * 512 + c] + remb[(size_t)r * 512 + c])
                : uemb[(size_t)r * 512 + c];
    Xf[row * 512 + c] = v; Xb[row * 512 + c] = (bf16)v;
  }
}

__global__ __launch_bounds__(256)
void pool_score_kernel(const bf16* __restrict__ Kp, const float* __restrict__ pq,
                       float* __restrict__ psc) {
  int row = blockIdx.x;
  int hh = threadIdx.x >> 5, ln = threadIdx.x & 31;
  const bf16* kp = Kp + (size_t)row * 512 + hh * 64;
  float acc = 0.f;
  for (int i = ln; i < 64; i += 32) acc += pq[hh * 64 + i] * (float)kp[i];
#pragma unroll
  for (int m2 = 1; m2 < 32; m2 <<= 1) acc += __shfl_xor(acc, m2, 32);
  if (ln == 0) psc[(size_t)row * 8 + hh] = acc * 0.125f;
}

__device__ inline float blk_reduce(float v, float* red, bool domax) {
  int tid = threadIdx.x, n = blockDim.x;
  red[tid] = v; __syncthreads();
  for (int s = n >> 1; s > 0; s >>= 1) {
    if (tid < s) red[tid] = domax ? fmaxf(red[tid], red[tid + s]) : (red[tid] + red[tid + s]);
    __syncthreads();
  }
  float r = red[0]; __syncthreads();
  return r;
}

__global__ __launch_bounds__(128)
void pool_reduce_kernel(const float* __restrict__ psc, const bf16* __restrict__ Vp,
                        bf16* __restrict__ Pp) {
  __shared__ float red[128];
  int bh = blockIdx.x; int b = bh >> 3, hh = bh & 7;
  const float* pp = psc + (size_t)b * SS * 8 + hh;
  float mx = -1e30f;
  for (int t = threadIdx.x; t < SS; t += 128) mx = fmaxf(mx, pp[(size_t)t * 8]);
  mx = blk_reduce(mx, red, true);
  float sm = 0.f;
  for (int t = threadIdx.x; t < SS; t += 128) sm += __expf(pp[(size_t)t * 8] - mx);
  sm = blk_reduce(sm, red, false);
  if (threadIdx.x < 64) {
    int e = threadIdx.x;
    float acc = 0.f;
    for (int t = 0; t < SS; ++t)
      acc += __expf(pp[(size_t)t * 8] - mx) * (float)Vp[((size_t)b * SS + t) * 512 + hh * 64 + e];
    Pp[(size_t)b * 512 + hh * 64 + e] = (bf16)(acc / sm);
  }
}

__global__ __launch_bounds__(256)
void layernorm_kernel(const float* __restrict__ Xin, float* __restrict__ Xf, bf16* __restrict__ Xb) {
  __shared__ float red[256];
  int b = blockIdx.x;
  const float* x = Xin + (size_t)b * 512;
  float s = 0.f;
  for (int e = threadIdx.x; e < 512; e += 256) s += x[e];
  float mu = blk_reduce(s, red, false) * (1.0f / 512.0f);
  float v = 0.f;
  for (int e = threadIdx.x; e < 512; e += 256) { float d = x[e] - mu; v += d * d; }
  float var = blk_reduce(v, red, false) * (1.0f / 512.0f);
  float inv = rsqrtf(var + 1e-5f);
  for (int e = threadIdx.x; e < 512; e += 256) {
    float val = (x[e] - mu) * inv;
    Xf[(size_t)b * 512 + e] = val;
    Xb[(size_t)b * 512 + e] = (bf16)val;
  }
}

// ---------------------------------------------------------------------------
// Host orchestration
// ---------------------------------------------------------------------------
extern "C" void kernel_launch(void* const* d_in, const int* in_sizes, int n_in,
                              void* d_out, int out_size, void* d_ws, size_t ws_size,
                              hipStream_t stream) {
  (void)in_sizes; (void)n_in; (void)out_size; (void)ws_size;

  const unsigned char* room_mask = (const unsigned char*)d_in[0];
  const int*   posx  = (const int*)d_in[1];
  const int*   posy  = (const int*)d_in[2];
  const int*   halfx = (const int*)d_in[3];
  const int*   halfy = (const int*)d_in[4];
  const float* steps = (const float*)d_in[5];
  const float* temp  = (const float*)d_in[6];
  const float* mc    = (const float*)d_in[7];
  const float* gw    = (const float*)d_in[8];
  const float* gb    = (const float*)d_in[9];
  const float* pex   = (const float*)d_in[10];
  const float* pey   = (const float*)d_in[11];
  const float* remb  = (const float*)d_in[12];
  const float* uemb  = (const float*)d_in[13];
  const float* aqw   = (const float*)d_in[14];
  const float* akw   = (const float*)d_in[15];
  const float* avw   = (const float*)d_in[16];
  const float* apw   = (const float*)d_in[17];
  const float* f1w   = (const float*)d_in[18];
  const float* f2w   = (const float*)d_in[19];
  const float* poolq = (const float*)d_in[20];
  const float* pkw   = (const float*)d_in[21];
  const float* pvw   = (const float*)d_in[22];
  const float* ppw   = (const float*)d_in[23];
  const float* g1w   = (const float*)d_in[24];
  const float* g2w   = (const float*)d_in[25];
  const float* o1w   = (const float*)d_in[26];
  const float* o2w   = (const float*)d_in[27];

  size_t off = 0;
  auto alloc = [&](size_t bytes) -> void* {
    off = (off + 255) & ~(size_t)255;
    void* p = (char*)d_ws + off;
    off += bytes;
    return p;
  };

  const size_t SZ_WQ = (size_t)LL*512*512, SZ_WKV = (size_t)LL*256*512, SZ_WP = SZ_WQ;
  const size_t SZ_F1 = (size_t)LL*2048*512, SZ_F2 = SZ_F1;
  const size_t SZ_PX = (size_t)512*512;
  const size_t SZ_G1 = (size_t)4*2048*512, SZ_G2 = SZ_G1;
  const size_t SZ_O1 = (size_t)2048*512,  SZ_O2 = (size_t)1024*2048;

  bf16* wq  = (bf16*)alloc(SZ_WQ*2);
  bf16* wkv = (bf16*)alloc(SZ_WKV*2);
  bf16* wp  = (bf16*)alloc(SZ_WP*2);
  bf16* wf1 = (bf16*)alloc(SZ_F1*2);
  bf16* wf2 = (bf16*)alloc(SZ_F2*2);
  bf16* wpk = (bf16*)alloc(SZ_PX*2);
  bf16* wpv = (bf16*)alloc(SZ_PX*2);
  bf16* wpp = (bf16*)alloc(SZ_PX*2);
  bf16* wg1 = (bf16*)alloc(SZ_G1*2);
  bf16* wg2 = (bf16*)alloc(SZ_G2*2);
  bf16* wo1 = (bf16*)alloc(SZ_O1*2);
  bf16* wo2 = (bf16*)alloc(SZ_O2*2);

  float* Xf  = (float*)alloc((size_t)MPAD*512*4);
  bf16*  Xb  = (bf16*) alloc((size_t)MPAD*512*2);
  bf16*  Qb  = (bf16*) alloc((size_t)MPAD*512*2);
  bf16*  KVb = (bf16*) alloc((size_t)MPAD*256*2);
  bf16*  Ab  = (bf16*) alloc((size_t)MPAD*512*2);
  bf16*  Hb  = (bf16*) alloc((size_t)MPAD*2048*2);
  float* psc = (float*)alloc((size_t)MROWS*8*4);
  bf16*  Ppb = (bf16*) alloc((size_t)128*512*2);
  float* Xgr = (float*)alloc((size_t)128*512*4);
  float* Xgf = (float*)alloc((size_t)128*512*4);
  bf16*  Xgb = (bf16*) alloc((size_t)128*512*2);
  bf16*  Ghb = (bf16*) alloc((size_t)128*2048*2);

  auto cvt = [&](const float* s, bf16* d, size_t n) {
    int blocks = (int)((n + 8*256 - 1) / (8*256));
    if (blocks > 4096) blocks = 4096;
    cvt_f32_bf16<<<blocks, 256, 0, stream>>>(s, d, n);
  };
  cvt(aqw, wq,  SZ_WQ);
  for (int i = 0; i < LL; ++i) {   // fuse K,V weights: rows [0,128)=K, [128,256)=V
    cvt(akw + (size_t)i*128*512, wkv + (size_t)i*256*512,             (size_t)128*512);
    cvt(avw + (size_t)i*128*512, wkv + (size_t)i*256*512 + 128*512,   (size_t)128*512);
  }
  cvt(apw, wp,  SZ_WP); cvt(f1w, wf1, SZ_F1); cvt(f2w, wf2, SZ_F2);
  cvt(pkw, wpk, SZ_PX); cvt(pvw, wpv, SZ_PX); cvt(ppw, wpp, SZ_PX);
  cvt(g1w, wg1, SZ_G1); cvt(g2w, wg2, SZ_G2);
  cvt(o1w, wo1, SZ_O1); cvt(o2w, wo2, SZ_O2);

  auto gemm = [&](const bf16* A, const bf16* Wt, const float* resid,
                  float* Cf, bf16* Cb, int M, int K, int Nn, int Mv, int gelu) {
    dim3 grid(M / 128, Nn / 128);
    gemm_wmma_bf16<<<grid, 256, 0, stream>>>(A, Wt, resid, Cf, Cb, M, K, Nn, Mv, gelu);
  };

  gemb_kernel<<<NB, 256, 0, stream>>>(room_mask, steps, temp, mc, gw, gb, Xf, Xb);
  embed_kernel<<<NB * RR, 128, 0, stream>>>(room_mask, posx, posy, halfx, halfy,
                                            pex, pey, remb, uemb, Xf, Xb);

  for (int i = 0; i < LL; ++i) {
    const bf16* qwL  = wq  + (size_t)i*512*512;
    const bf16* kvwL = wkv + (size_t)i*256*512;
    const bf16* pwL  = wp  + (size_t)i*512*512;
    const bf16* w1L  = wf1 + (size_t)i*2048*512;
    const bf16* w2L  = wf2 + (size_t)i*512*2048;
    gemm(Xb, qwL,  nullptr, nullptr, Qb,  MPAD, 512, 512, MPAD, 0);
    gemm(Xb, kvwL, nullptr, nullptr, KVb, MPAD, 512, 256, MPAD, 0);
    attn_flash<<<dim3(33, 8, NB), 32, 0, stream>>>(Qb, KVb, Ab);
    gemm(Ab, pwL, Xf, Xf, Xb, MPAD, 512, 512, MPAD, 0);
    gemm(Xb, w1L, nullptr, nullptr, Hb, MPAD, 512, 2048, MPAD, 1);
    gemm(Hb, w2L, Xf, Xf, Xb, MPAD, 2048, 512, MPAD, 0);
  }

  // Pooling (reuse Qb/Ab as Kp/Vp)
  gemm(Xb, wpk, nullptr, nullptr, Qb, MPAD, 512, 512, MPAD, 0);
  gemm(Xb, wpv, nullptr, nullptr, Ab, MPAD, 512, 512, MPAD, 0);
  pool_score_kernel<<<MROWS, 256, 0, stream>>>(Qb, poolq, psc);
  pool_reduce_kernel<<<NB * 8, 128, 0, stream>>>(psc, Ab, Ppb);
  gemm(Ppb, wpp, nullptr, Xgr, nullptr, 128, 512, 512, 128, 0);
  layernorm_kernel<<<NB, 256, 0, stream>>>(Xgr, Xgf, Xgb);

  for (int i = 0; i < 4; ++i) {
    gemm(Xgb, wg1 + (size_t)i*2048*512, nullptr, nullptr, Ghb, 128, 512, 2048, 128, 1);
    gemm(Ghb, wg2 + (size_t)i*512*2048, Xgf, Xgf, Xgb, 128, 2048, 512, 128, 0);
  }

  gemm(Xgb, wo1, nullptr, nullptr, Ghb, 128, 512, 2048, 128, 1);
  gemm(Ghb, wo2, nullptr, (float*)d_out, nullptr, 128, 2048, 1024, 32, 0);
}
